// LayoutAttention_56908316672230
// MI455X (gfx1250) — matrix operand order repaired and checked by reading, not compiled, not run on hardware
//
#include <hip/hip_runtime.h>
#include <hip/hip_bf16.h>

// Problem constants (from the reference)
#define BB   2
#define NN   2304
#define DMD  1024
#define NH   8
#define DHD  128
#define NP   4

typedef __bf16 bf16_t;
typedef __attribute__((ext_vector_type(16))) __bf16 v16bf;
typedef __attribute__((ext_vector_type(8)))  float  v8f;

constexpr int GM = 4608;       // B*N
constexpr int GK = 1024;
constexpr int GN = 1024;
constexpr int KSTEP  = 32;
constexpr int KITERS = GK / KSTEP;    // 32

__device__ __forceinline__ unsigned short f2bf_u16(float f) {
  unsigned u = __builtin_bit_cast(unsigned, f);
  unsigned r = u + 0x7FFFu + ((u >> 16) & 1u);   // round-to-nearest-even
  return (unsigned short)(r >> 16);
}
__device__ __forceinline__ bf16_t f2bf(float f) {
  unsigned short h = f2bf_u16(f);
  return __builtin_bit_cast(bf16_t, h);
}

// ---- gfx1250 async global->LDS copy (ASYNCcnt-tracked, ISA 15.18.3) -----
__device__ __forceinline__ void async_g2l_b128(void* lds, const void* gptr) {
  unsigned loff = (unsigned)(size_t)lds;          // low 32 bits of generic = LDS offset
  unsigned long long ga = (unsigned long long)(size_t)gptr;
  asm volatile("global_load_async_to_lds_b128 %0, %1, off"
               :: "v"(loff), "v"(ga) : "memory");
}
template <int N>
__device__ __forceinline__ void wait_async_le() {
  asm volatile("s_wait_asynccnt %0" :: "n"(N) : "memory");
}

// ---- WMMA fragment loaders (CDNA5 ISA 7.12.2, wave32) -------------------
// A fragment: 16x32 (MxK) bf16, row-major source; 2 contiguous 8-elem runs.
__device__ __forceinline__ v16bf load_a_frag(const bf16_t* src, int ld, unsigned lane) {
  const int m  = lane & 15;
  const int hi = (lane >> 4) & 1;
  v16bf a;
#pragma unroll
  for (int e = 0; e < 8; ++e) {
    a[e]     = src[m * ld + hi * 8 + e];           // K = hi*8 + 0..7
    a[e + 8] = src[m * ld + 16 + hi * 8 + e];      // K = 16 + hi*8 + 0..7
  }
  return a;
}

// B fragment 32x16 (KxN), source stored N-major: B[k][n] = src[n*ld + k].
__device__ __forceinline__ v16bf load_bT_frag(const bf16_t* src, int ld, unsigned lane) {
  const int n  = lane & 15;
  const int hi = (lane >> 4) & 1;
  v16bf b;
#pragma unroll
  for (int e = 0; e < 16; ++e)
    b[e] = src[n * ld + (e < 8 ? hi * 8 + e : 16 + hi * 8 + (e - 8))];
  return b;
}

// ---- elementwise f32 -> bf16, 8 elems/thread, vectorized ----------------
__global__ void __launch_bounds__(256) f32_to_bf16_vec8(
    const float* __restrict__ src, bf16_t* __restrict__ dst, int n8) {
  int i = blockIdx.x * 256 + threadIdx.x;
  if (i >= n8) return;
  const float4* s = reinterpret_cast<const float4*>(src) + (size_t)i * 2;
  float4 a = s[0], b = s[1];
  union { uint4 v; unsigned short h[8]; } pk;
  pk.h[0] = f2bf_u16(a.x); pk.h[1] = f2bf_u16(a.y);
  pk.h[2] = f2bf_u16(a.z); pk.h[3] = f2bf_u16(a.w);
  pk.h[4] = f2bf_u16(b.x); pk.h[5] = f2bf_u16(b.y);
  pk.h[6] = f2bf_u16(b.z); pk.h[7] = f2bf_u16(b.w);
  reinterpret_cast<uint4*>(dst)[i] = pk.v;
}

// ---- tiled transpose + convert: dst[N x K] = bf16(src[K x N]) -----------
__global__ void __launch_bounds__(256) transpose_f32_to_bf16(
    const float* __restrict__ src, bf16_t* __restrict__ dst) {
  __shared__ float tile[32][33];
  const int k0 = blockIdx.x * 32;
  const int n0 = blockIdx.y * 32;
  const int tx = threadIdx.x & 31;
  const int ty = threadIdx.x >> 5;            // 0..7
#pragma unroll
  for (int i = 0; i < 32; i += 8)
    tile[ty + i][tx] = src[(size_t)(k0 + ty + i) * DMD + n0 + tx];
  __syncthreads();
#pragma unroll
  for (int i = 0; i < 32; i += 8)
    dst[(size_t)(n0 + ty + i) * DMD + k0 + tx] = f2bf(tile[tx][ty + i]);
}

// ---- per-token phase bitmask: mask(i,j) == ((pm[i]&pm[j])==0) -----------
__global__ void __launch_bounds__(256) phase_mask_kernel(
    const float* __restrict__ gmask, unsigned char* __restrict__ pmv) {
  int idx = blockIdx.x * 256 + threadIdx.x;     // over B*N
  if (idx >= BB * NN) return;
  int b = idx / NN, i = idx % NN;
  unsigned m = 0;
#pragma unroll
  for (int p = 0; p < NP; ++p)
    if (gmask[((size_t)b * NP + p) * NN + i] > 0.5f) m |= (1u << p);
  pmv[idx] = (unsigned char)m;
}

// ---- bf16 WMMA GEMM core: Out = A[GM,GK] @ Wt^T (+bias) -----------------
// Wt is [GN x GK] (pre-transposed). 8 waves; wave tile 32x64; block 256x64.
// Double-buffered async staging: tile t+1 streams in while tile t computes.
__device__ __forceinline__ void gemm_core(
    const bf16_t* __restrict__ A, const bf16_t* __restrict__ Wt,
    const float* __restrict__ bias, float* __restrict__ outF,
    bf16_t* __restrict__ outB, int transOut,
    bf16_t* As /* [2][256*KSTEP] */, bf16_t* Bs /* [2][64*KSTEP] */) {
  const unsigned t    = threadIdx.x;
  const unsigned lane = t & 31;
  const unsigned wid  = t >> 5;
  const int m0 = blockIdx.x * 256;
  const int n0 = blockIdx.y * 64;

  auto stage = [&](int kt, int buf) {
    const int k0 = kt * KSTEP;
    bf16_t* as = As + buf * (256 * KSTEP);
    bf16_t* bs = Bs + buf * (64 * KSTEP);
    // A tile 256x32: 1024 b128 transfers, 4 per thread
#pragma unroll
    for (int i = 0; i < 4; ++i) {
      unsigned u = t + i * 256;
      int row = u >> 2, seg = u & 3;
      async_g2l_b128(&as[row * KSTEP + seg * 8],
                     A + (size_t)(m0 + row) * GK + k0 + seg * 8);
    }
    // B tile (n-major) 64x32: 256 b128 transfers, 1 per thread
    {
      int row = t >> 2, seg = t & 3;
      async_g2l_b128(&bs[row * KSTEP + seg * 8],
                     Wt + (size_t)(n0 + row) * GK + k0 + seg * 8);
    }
  };

  v8f acc[2][4] = {};
  stage(0, 0);
  for (int kt = 0; kt < KITERS; ++kt) {
    const int cur = kt & 1;
    if (kt + 1 < KITERS) { stage(kt + 1, cur ^ 1); wait_async_le<5>(); }
    else                 { wait_async_le<0>(); }
    __syncthreads();                       // tile kt visible to all waves

    const bf16_t* as = As + cur * (256 * KSTEP);
    const bf16_t* bs = Bs + cur * (64 * KSTEP);
    v16bf a0 = load_a_frag(&as[(wid * 32) * KSTEP], KSTEP, lane);
    v16bf a1 = load_a_frag(&as[(wid * 32 + 16) * KSTEP], KSTEP, lane);
#pragma unroll
    for (int nt = 0; nt < 4; ++nt) {
      v16bf b = load_bT_frag(&bs[(nt * 16) * KSTEP], KSTEP, lane);
      acc[0][nt] = __builtin_amdgcn_wmma_f32_16x16x32_bf16(
          false, a0, false, b, (short)0, acc[0][nt], false, false);
      acc[1][nt] = __builtin_amdgcn_wmma_f32_16x16x32_bf16(
          false, a1, false, b, (short)0, acc[1][nt], false, false);
    }
    __syncthreads();                       // all reads of buf[cur] done (WAR)
  }

  const int n  = lane & 15;
  const int hi = (lane >> 4) & 1;
#pragma unroll
  for (int mi = 0; mi < 2; ++mi) {
    const int mbase = m0 + (int)wid * 32 + mi * 16;
    if (outB && transOut) {
      // transposed store: r=0..7 consecutive -> one uint4 per nt
#pragma unroll
      for (int nt = 0; nt < 4; ++nt) {
        union { uint4 v; unsigned short h[8]; } pk;
#pragma unroll
        for (int r = 0; r < 8; ++r) pk.h[r] = f2bf_u16(acc[mi][nt][r]);
        int col = n0 + nt * 16 + n;
        size_t base = (size_t)col * GM + mbase + hi * 8;
        *reinterpret_cast<uint4*>(outB + base) = pk.v;
      }
    } else {
#pragma unroll
      for (int nt = 0; nt < 4; ++nt) {
#pragma unroll
        for (int r = 0; r < 8; ++r) {
          int row = mbase + r + hi * 8;
          int col = n0 + nt * 16 + n;
          float v = acc[mi][nt][r];
          if (outB) outB[(size_t)row * GN + col] = f2bf(v);
          else      outF[(size_t)row * GN + col] = v + (bias ? bias[col] : 0.0f);
        }
      }
    }
  }
}

// Fused Q/K/V projection: blockIdx.z selects the weight; V written transposed.
__global__ void __launch_bounds__(256) gemm_qkv_wmma(
    const bf16_t* __restrict__ xb,
    const bf16_t* __restrict__ Wqt, const bf16_t* __restrict__ Wkt,
    const bf16_t* __restrict__ Wvt,
    bf16_t* __restrict__ Qb, bf16_t* __restrict__ Kb, bf16_t* __restrict__ Vt) {
  __shared__ alignas(16) bf16_t As[2][256 * KSTEP];   // 2 x 16 KB
  __shared__ alignas(16) bf16_t Bs[2][64 * KSTEP];    // 2 x  4 KB
  const int z = blockIdx.z;
  const bf16_t* Wt = (z == 0) ? Wqt : (z == 1) ? Wkt : Wvt;
  bf16_t* outB     = (z == 0) ? Qb  : (z == 1) ? Kb  : Vt;
  gemm_core(xb, Wt, nullptr, nullptr, outB, z == 2, &As[0][0], &Bs[0][0]);
}

// Output projection with bias, f32 result.
__global__ void __launch_bounds__(256) gemm_out_wmma(
    const bf16_t* __restrict__ Ob, const bf16_t* __restrict__ Wot,
    const float* __restrict__ bias, float* __restrict__ out) {
  __shared__ alignas(16) bf16_t As[2][256 * KSTEP];
  __shared__ alignas(16) bf16_t Bs[2][64 * KSTEP];
  gemm_core(Ob, Wot, bias, out, nullptr, 0, &As[0][0], &Bs[0][0]);
}

// ---- flash attention: 4 waves/block share (b,h); each wave = 16 q-rows ---
// Qb,Kb row-major [B*N, DM]; Vt transposed [DM, B*N]. Double-buffered K/V.
__global__ void __launch_bounds__(128) flash_attn_wmma(
    const bf16_t* __restrict__ Qb, const bf16_t* __restrict__ Kb,
    const bf16_t* __restrict__ Vt, const unsigned char* __restrict__ pmv,
    bf16_t* __restrict__ Ob) {
  __shared__ alignas(16) bf16_t Ks[2][32 * 128];    // [j][d], 2 x 8 KB
  __shared__ alignas(16) bf16_t Vsw[2][128 * 32];   // [d][j], 2 x 8 KB
  __shared__ alignas(16) bf16_t Ps[4][16 * 32];     // 4 KB

  const unsigned t    = threadIdx.x;
  const unsigned lane = t & 31;
  const unsigned wid  = t >> 5;
  const int b  = blockIdx.z;
  const int h  = blockIdx.y;
  const int i0 = (blockIdx.x * 4 + (int)wid) * 16;   // 16-row query tile
  const float scale = 0.088388347648318447f;         // 1/sqrt(128)

  const int nlo = lane & 15;
  const int hi  = (lane >> 4) & 1;

  const size_t kbase  = ((size_t)b * NN) * DMD + (size_t)h * DHD;     // K row-major
  const size_t vtbase = (size_t)h * DHD * (BB * NN) + (size_t)b * NN; // Vt: row=d, col=token

  auto stageKV = [&](int jt, int buf) {
    const int j0 = jt * 32;
    bf16_t* ks = &Ks[buf][0];
    bf16_t* vs = &Vsw[buf][0];
#pragma unroll
    for (int i = 0; i < 4; ++i) {
      unsigned u = t + i * 128;
      int row = u >> 4, seg = u & 15;
      async_g2l_b128(&ks[row * 128 + seg * 8],
                     Kb + kbase + (size_t)(j0 + row) * DMD + seg * 8);
    }
#pragma unroll
    for (int i = 0; i < 4; ++i) {
      unsigned u = t + i * 128;
      int d = u >> 2, seg = u & 3;
      async_g2l_b128(&vs[d * 32 + seg * 8],
                     Vt + vtbase + (size_t)d * (BB * NN) + j0 + seg * 8);
    }
  };

  // Q fragments (4 chunks of 32 over DH=128) kept in VGPRs; b128 global loads
  v16bf qf[4];
  const bf16_t* qbase = Qb + ((size_t)b * NN + i0) * DMD + h * DHD;
#pragma unroll
  for (int kc = 0; kc < 4; ++kc) qf[kc] = load_a_frag(qbase + kc * 32, DMD, lane);

  unsigned pmi[8];
#pragma unroll
  for (int r = 0; r < 8; ++r) pmi[r] = pmv[(size_t)b * NN + i0 + r + hi * 8];

  float rmax[8], rsum[8];
#pragma unroll
  for (int r = 0; r < 8; ++r) { rmax[r] = -3.0e38f; rsum[r] = 0.0f; }
  v8f oacc[8] = {};

  constexpr int JT = NN / 32;   // 72
  stageKV(0, 0);
  for (int jt = 0; jt < JT; ++jt) {
    const int cur = jt & 1;
    if (jt + 1 < JT) { stageKV(jt + 1, cur ^ 1); wait_async_le<8>(); }
    else             { wait_async_le<0>(); }
    __syncthreads();

    const int j0 = jt * 32;
    const bf16_t* ks = &Ks[cur][0];
    const bf16_t* vs = &Vsw[cur][0];

    // S = Q·K^T : two 16x16 tiles over a 32-key block (B-frags contiguous)
    v8f s0 = {}, s1 = {};
#pragma unroll
    for (int kc = 0; kc < 4; ++kc) {
      v16bf b0 = load_bT_frag(&ks[0 * 128 + kc * 32], 128, lane);
      s0 = __builtin_amdgcn_wmma_f32_16x16x32_bf16(false, qf[kc], false, b0, (short)0, s0, false, false);
      v16bf b1 = load_bT_frag(&ks[16 * 128 + kc * 32], 128, lane);
      s1 = __builtin_amdgcn_wmma_f32_16x16x32_bf16(false, qf[kc], false, b1, (short)0, s1, false, false);
    }

    const unsigned pmj0 = pmv[(size_t)b * NN + j0 + nlo];
    const unsigned pmj1 = pmv[(size_t)b * NN + j0 + 16 + nlo];

    // online softmax (rows live across 16-lane halves; width-16 shuffles)
#pragma unroll
    for (int r = 0; r < 8; ++r) {
      float v0 = s0[r] * scale;
      float v1 = s1[r] * scale;
      if ((pmi[r] & pmj0) == 0u) v0 = -3.0e38f;
      if ((pmi[r] & pmj1) == 0u) v1 = -3.0e38f;
      float mloc = fmaxf(v0, v1);
#pragma unroll
      for (int xm = 1; xm < 16; xm <<= 1) mloc = fmaxf(mloc, __shfl_xor(mloc, xm, 16));
      float mnew = fmaxf(rmax[r], mloc);
      float corr = __expf(rmax[r] - mnew);
      rsum[r] *= corr;
#pragma unroll
      for (int dc = 0; dc < 8; ++dc) oacc[dc][r] *= corr;
      float p0 = __expf(v0 - mnew);
      float p1 = __expf(v1 - mnew);
      rsum[r] += p0 + p1;           // per-lane partial; reduced once at the end
      rmax[r] = mnew;
      int row = r + hi * 8;
      Ps[wid][row * 32 + nlo]      = f2bf(p0);   // reshape C-layout -> A-layout via LDS
      Ps[wid][row * 32 + 16 + nlo] = f2bf(p1);
    }
    __syncthreads();

    // O += P · V  (A = P 16x32; B-frags from d-major Vsw are contiguous)
    v16bf pa = load_a_frag(&Ps[wid][0], 32, lane);
#pragma unroll
    for (int dc = 0; dc < 8; ++dc) {
      v16bf bv = load_bT_frag(&vs[(dc * 16) * 32], 32, lane);
      oacc[dc] = __builtin_amdgcn_wmma_f32_16x16x32_bf16(
          false, pa, false, bv, (short)0, oacc[dc], false, false);
    }
    __syncthreads();                // all reads of buf[cur] done (WAR)
  }

  // normalize and store O (bf16, [B,N,DM] layout for the output projection)
#pragma unroll
  for (int r = 0; r < 8; ++r) {
    float s = rsum[r];
#pragma unroll
    for (int xm = 1; xm < 16; xm <<= 1) s += __shfl_xor(s, xm, 16);
    float inv = 1.0f / s;
    int row_g = i0 + r + hi * 8;
    bf16_t* obase = Ob + ((size_t)b * NN + row_g) * DMD + h * DHD;
#pragma unroll
    for (int dc = 0; dc < 8; ++dc)
      obase[dc * 16 + nlo] = f2bf(oacc[dc][r] * inv);
  }
}

// -------------------------------------------------------------------------
extern "C" void kernel_launch(void* const* d_in, const int* in_sizes, int n_in,
                              void* d_out, int out_size, void* d_ws, size_t ws_size,
                              hipStream_t stream) {
  (void)in_sizes; (void)n_in; (void)out_size; (void)ws_size;
  const float* x  = (const float*)d_in[0];
  const float* gm = (const float*)d_in[1];
  const float* Wq = (const float*)d_in[2];
  const float* Wk = (const float*)d_in[3];
  const float* Wv = (const float*)d_in[4];
  const float* Wo = (const float*)d_in[5];
  const float* bo = (const float*)d_in[6];
  float* out = (float*)d_out;
  char* ws = (char*)d_ws;

  const size_t BN      = (size_t)BB * NN;                 // 4608
  const size_t xbBytes = BN * DMD * sizeof(bf16_t);       // 9,437,184
  const size_t wBytes  = (size_t)DMD * DMD * sizeof(bf16_t);

  bf16_t* xb  = (bf16_t*)(ws);                      // reused as attention output
  bf16_t* Qb  = (bf16_t*)(ws + 1 * xbBytes);
  bf16_t* Kb  = (bf16_t*)(ws + 2 * xbBytes);
  bf16_t* Vt  = (bf16_t*)(ws + 3 * xbBytes);        // transposed V [DM, B*N]
  bf16_t* Wqt = (bf16_t*)(ws + 4 * xbBytes);        // transposed weights [N,K]
  bf16_t* Wkt = (bf16_t*)(ws + 4 * xbBytes + 1 * wBytes);
  bf16_t* Wvt = (bf16_t*)(ws + 4 * xbBytes + 2 * wBytes);
  bf16_t* Wot = (bf16_t*)(ws + 4 * xbBytes + 3 * wBytes);
  unsigned char* pmv = (unsigned char*)(ws + 4 * xbBytes + 4 * wBytes);

  const int nx8 = (int)(BN * DMD / 8);
  f32_to_bf16_vec8<<<(nx8 + 255) / 256, 256, 0, stream>>>(x, xb, nx8);
  dim3 tg(DMD / 32, DMD / 32);
  transpose_f32_to_bf16<<<tg, 256, 0, stream>>>(Wq, Wqt);
  transpose_f32_to_bf16<<<tg, 256, 0, stream>>>(Wk, Wkt);
  transpose_f32_to_bf16<<<tg, 256, 0, stream>>>(Wv, Wvt);
  transpose_f32_to_bf16<<<tg, 256, 0, stream>>>(Wo, Wot);
  phase_mask_kernel<<<((int)BN + 255) / 256, 256, 0, stream>>>(gm, pmv);

  gemm_qkv_wmma<<<dim3(GM / 256, GN / 64, 3), 256, 0, stream>>>(
      xb, Wqt, Wkt, Wvt, Qb, Kb, Vt);

  flash_attn_wmma<<<dim3(36, NH, BB), 128, 0, stream>>>(Qb, Kb, Vt, pmv, xb);

  gemm_out_wmma<<<dim3(GM / 256, GN / 64), 256, 0, stream>>>(xb, Wot, bo, out);
}
